// TrasformerEncoder_32091995635852
// MI455X (gfx1250) — matrix operand order repaired and checked by reading, not compiled, not run on hardware
//
#include <hip/hip_runtime.h>

// ---------------------------------------------------------------------------
// Transformer encoder block for MI455X (gfx1250), wave32, WMMA bf16 path.
// All GEMMs run through v_wmma_f32_16x16x32_bf16 (bf16 A/B, f32 accumulate).
// Wave tile 32x64 (2 A-frags x 4 B-frags -> 8 WMMA / 12 b128 loads per K=32).
// ---------------------------------------------------------------------------

typedef __attribute__((ext_vector_type(8)))  float  v8f;
typedef __attribute__((ext_vector_type(16))) __bf16 v16bf;
typedef __attribute__((ext_vector_type(8)))  __bf16 v8bf;

constexpr int TE_B = 8;
constexpr int TE_S = 1024;
constexpr int TE_D = 768;
constexpr int TE_H = 3072;

__device__ __forceinline__ float te_gelu(float x) {
    return 0.5f * x * (1.0f + erff(x * 0.70710678118654752f));
}

// ---------------------------------------------------------------------------
// Weight convert + transpose: W[R,C] f32 -> Wt[C,R] bf16
// ---------------------------------------------------------------------------
__global__ __launch_bounds__(256)
void te_cvt_t(const float* __restrict__ W, __bf16* __restrict__ Wt, int R, int C) {
    long idx = (long)blockIdx.x * 256 + threadIdx.x;
    if (idx < (long)R * C) {
        int r = (int)(idx / C);
        int c = (int)(idx % C);
        Wt[(long)c * R + r] = (__bf16)W[idx];
    }
}

__global__ void te_zero(float* __restrict__ p, int n) {
    int i = blockIdx.x * blockDim.x + threadIdx.x;
    if (i < n) p[i] = 0.0f;
}

// ---------------------------------------------------------------------------
// Embedding lookup + sinusoidal positional encoding; writes f32 + bf16 copy.
// ---------------------------------------------------------------------------
__global__ __launch_bounds__(256)
void te_embed(const int* __restrict__ x, const float* __restrict__ emb,
              float* __restrict__ y, __bf16* __restrict__ ybf) {
    const long row = blockIdx.x;                 // b*S + s
    const int  s   = (int)(row & (TE_S - 1));
    const int  tok = x[row];
    const float* e = emb + (long)tok * TE_D;
    float*  yo = y   + row * (long)TE_D;
    __bf16* yb = ybf + row * (long)TE_D;
    for (int d = threadIdx.x; d < TE_D; d += 256) {
        // ang = pos * 10000^(-d/768) = pos * exp(-d * ln(10000)/768)
        float ang = (float)s * __expf(-(float)d * 0.01199263069f);
        float pe  = (d & 1) ? __cosf(ang) : __sinf(ang);
        float v   = e[d] + pe;
        yo[d] = v;
        yb[d] = (__bf16)v;
    }
}

// ---------------------------------------------------------------------------
// WMMA GEMM:  C[M,N] = A[M,K] (bf16, row-major) x Bt[N,K] (bf16, row-major)
// Block: 256 threads = 8 waves; block tile 256(M) x 64(N); wave tile 32x64.
// blockIdx.z batches via element strides sA/sB/sC.
// MODE 0: C f32 row-major   MODE 1: C bf16 row-major
// MODE 2: C bf16 transposed per batch: C[n*ldc + m]
// ---------------------------------------------------------------------------
template <int MODE, bool BIAS>
__global__ __launch_bounds__(256)
void te_gemm(const __bf16* __restrict__ A, long sA, int lda,
             const __bf16* __restrict__ Bt, long sB, int ldb,
             void* __restrict__ Cv, long sC, int ldc,
             const float* __restrict__ bias, int K) {
    const int lane = threadIdx.x & 31;
    const int wave = threadIdx.x >> 5;
    const int hi   = lane >> 4;     // half-wave select
    const int l16  = lane & 15;
    const long b   = blockIdx.z;

    A  += b * sA;
    Bt += b * sB;

    const int mBase = blockIdx.y * 256 + wave * 32;
    const int nBase = blockIdx.x * 64;

    // A fragment rows (16x32 bf16 each): lanes 0-15 hold K +0..15,
    // lanes 16-31 hold K +16..31 (contiguous, matching B).
    const __bf16* aRow0 = A + (long)(mBase + l16) * lda + (hi ? 16 : 0);
    const __bf16* aRow1 = aRow0 + (long)16 * lda;

    v8f acc[2][4];
#pragma unroll
    for (int u = 0; u < 2; ++u)
#pragma unroll
        for (int t = 0; t < 4; ++t)
#pragma unroll
            for (int r = 0; r < 8; ++r) acc[u][t][r] = 0.0f;

    for (int k = 0; k < K; k += 32) {
        union { v16bf v; v8bf h[2]; } a0, a1;
        a0.h[0] = *(const v8bf*)(aRow0 + k);
        a0.h[1] = *(const v8bf*)(aRow0 + k + 8);
        a1.h[0] = *(const v8bf*)(aRow1 + k);
        a1.h[1] = *(const v8bf*)(aRow1 + k + 8);
#pragma unroll
        for (int t = 0; t < 4; ++t) {
            // B fragment (32x16 bf16): lane = column, lanes 0-15 K k..k+15,
            // lanes 16-31 K k+16..k+31; contiguous along K since Bt is [N,K].
            const __bf16* bRow = Bt + (long)(nBase + t * 16 + l16) * ldb
                                    + k + (hi ? 16 : 0);
            v16bf bf = *(const v16bf*)bRow;
            acc[0][t] = __builtin_amdgcn_wmma_f32_16x16x32_bf16(
                false, a0.v, false, bf, (short)0, acc[0][t], false, false);
            acc[1][t] = __builtin_amdgcn_wmma_f32_16x16x32_bf16(
                false, a1.v, false, bf, (short)0, acc[1][t], false, false);
        }
    }

    // C layout: VGPR r, lanes 0-15 -> M=+r, lanes 16-31 -> M=+8+r; N=l16.
#pragma unroll
    for (int u = 0; u < 2; ++u) {
#pragma unroll
        for (int t = 0; t < 4; ++t) {
            const int nCol = nBase + t * 16 + l16;
            const float bv = BIAS ? bias[nCol] : 0.0f;
#pragma unroll
            for (int r = 0; r < 8; ++r) {
                const int m = mBase + u * 16 + (hi ? 8 : 0) + r;
                const float val = acc[u][t][r] + bv;
                if (MODE == 0)
                    ((float*)Cv)[b * sC + (long)m * ldc + nCol] = val;
                else if (MODE == 1)
                    ((__bf16*)Cv)[b * sC + (long)m * ldc + nCol] = (__bf16)val;
                else
                    ((__bf16*)Cv)[b * sC + (long)nCol * ldc + m] = (__bf16)val;
            }
        }
    }
}

// ---------------------------------------------------------------------------
// Row softmax over 1024 elements, f32 in -> bf16 out. One block per row.
// ---------------------------------------------------------------------------
__global__ __launch_bounds__(256)
void te_softmax(const float* __restrict__ sc, __bf16* __restrict__ P) {
    const long row = blockIdx.x;
    const float* src = sc + row * (long)TE_S;
    __bf16* dst = P + row * (long)TE_S;
    __shared__ float sh[256];
    float e[4];

    float lm = -3.4e38f;
#pragma unroll
    for (int j = 0; j < 4; ++j) lm = fmaxf(lm, src[threadIdx.x + j * 256]);
    sh[threadIdx.x] = lm;
    __syncthreads();
    for (int st = 128; st > 0; st >>= 1) {
        if (threadIdx.x < st) sh[threadIdx.x] = fmaxf(sh[threadIdx.x], sh[threadIdx.x + st]);
        __syncthreads();
    }
    const float m = sh[0];
    __syncthreads();

    float ls = 0.0f;
#pragma unroll
    for (int j = 0; j < 4; ++j) {
        e[j] = __expf(src[threadIdx.x + j * 256] - m);
        ls += e[j];
    }
    sh[threadIdx.x] = ls;
    __syncthreads();
    for (int st = 128; st > 0; st >>= 1) {
        if (threadIdx.x < st) sh[threadIdx.x] += sh[threadIdx.x + st];
        __syncthreads();
    }
    const float inv = 1.0f / sh[0];
#pragma unroll
    for (int j = 0; j < 4; ++j)
        dst[threadIdx.x + j * 256] = (__bf16)(e[j] * inv);
}

// ---------------------------------------------------------------------------
// layernorm2d over [S,D] per batch. Phase 1: partial sums -> atomics.
// x = base + (GELU ? gelu(add) : add)
// ---------------------------------------------------------------------------
template <bool GELU>
__global__ __launch_bounds__(256)
void te_ln_stats(const float* __restrict__ base, const float* __restrict__ add,
                 float* __restrict__ stats) {
    constexpr long NPER  = (long)TE_S * TE_D;   // 786432
    constexpr int  CHUNK = (int)(NPER / 64);    // 12288
    const int  b   = blockIdx.y;
    const long off = (long)b * NPER + (long)blockIdx.x * CHUNK;
    const float* pb = base + off;
    const float* pa = add + off;

    float s = 0.0f, q = 0.0f;
    for (int i = threadIdx.x; i < CHUNK; i += 256) {
        float t = pa[i];
        if (GELU) t = te_gelu(t);
        float xv = pb[i] + t;
        s += xv;
        q += xv * xv;
    }
    __shared__ float sh1[256], sh2[256];
    sh1[threadIdx.x] = s;
    sh2[threadIdx.x] = q;
    __syncthreads();
    for (int st = 128; st > 0; st >>= 1) {
        if (threadIdx.x < st) {
            sh1[threadIdx.x] += sh1[threadIdx.x + st];
            sh2[threadIdx.x] += sh2[threadIdx.x + st];
        }
        __syncthreads();
    }
    if (threadIdx.x == 0) {
        atomicAdd(&stats[2 * b], sh1[0]);
        atomicAdd(&stats[2 * b + 1], sh2[0]);
    }
}

// Phase 2: apply affine normalization; optional bf16 shadow copy for next GEMM.
template <bool GELU, bool BFOUT>
__global__ __launch_bounds__(256)
void te_ln_apply(const float* __restrict__ base, const float* __restrict__ add,
                 const float* __restrict__ w, const float* __restrict__ bb,
                 const float* __restrict__ stats,
                 float* __restrict__ out, __bf16* __restrict__ outbf) {
    constexpr long NPER = (long)TE_S * TE_D;
    const int b = blockIdx.y;
    const float mean = stats[2 * b] * (1.0f / (float)NPER);
    const float var  = stats[2 * b + 1] * (1.0f / (float)NPER) - mean * mean;
    const float inv  = rsqrtf(var + 1e-5f);
    const long boff  = (long)b * NPER;
    for (long i = (long)blockIdx.x * 256 + threadIdx.x; i < NPER;
         i += (long)gridDim.x * 256) {
        float t = add[boff + i];
        if (GELU) t = te_gelu(t);
        float xv = base[boff + i] + t;
        float r  = (xv - mean) * inv * w[i] + bb[i];
        out[boff + i] = r;
        if (BFOUT) outbf[boff + i] = (__bf16)r;
    }
}

// ---------------------------------------------------------------------------
// Host-side orchestration
// ---------------------------------------------------------------------------
extern "C" void kernel_launch(void* const* d_in, const int* in_sizes, int n_in,
                              void* d_out, int out_size, void* d_ws, size_t ws_size,
                              hipStream_t stream) {
    (void)in_sizes; (void)n_in; (void)out_size; (void)ws_size;

    const int*   x     = (const int*)  d_in[0];
    const float* emb   = (const float*)d_in[1];
    const float* Wq    = (const float*)d_in[2];
    const float* bq    = (const float*)d_in[3];
    const float* Wk    = (const float*)d_in[4];
    const float* bk    = (const float*)d_in[5];
    const float* Wv    = (const float*)d_in[6];
    const float* bv    = (const float*)d_in[7];
    const float* Wo    = (const float*)d_in[8];
    const float* bo    = (const float*)d_in[9];
    const float* W1    = (const float*)d_in[10];
    const float* b1    = (const float*)d_in[11];
    const float* W2    = (const float*)d_in[12];
    const float* b2    = (const float*)d_in[13];
    const float* ln1w  = (const float*)d_in[14];
    const float* ln1b  = (const float*)d_in[15];
    const float* ln2w  = (const float*)d_in[16];
    const float* ln2b  = (const float*)d_in[17];

    constexpr int  B = TE_B, S = TE_S, D = TE_D, H = TE_H;
    constexpr long BS = (long)B * S;

    // --- workspace carve-out (256B aligned sub-buffers) ---
    char* wp = (char*)d_ws;
    auto alloc = [&](size_t bytes) -> void* {
        void* p = (void*)wp;
        wp += (bytes + 255) & ~(size_t)255;
        return p;
    };
    float*  stats = (float*) alloc(64 * sizeof(float));        // ln1: [0..15], ln2: [16..31]
    __bf16* WqT   = (__bf16*)alloc((size_t)H * D * 2);         // [H,D]
    __bf16* WkT   = (__bf16*)alloc((size_t)H * D * 2);
    __bf16* WvT   = (__bf16*)alloc((size_t)H * D * 2);
    __bf16* WoT   = (__bf16*)alloc((size_t)D * H * 2);         // [D,H]
    __bf16* W1T   = (__bf16*)alloc((size_t)H * D * 2);
    __bf16* W2T   = (__bf16*)alloc((size_t)D * H * 2);
    __bf16* ybf   = (__bf16*)alloc((size_t)BS * D * 2);
    float*  y     = (float*) alloc((size_t)BS * D * 4);
    __bf16* q     = (__bf16*)alloc((size_t)BS * H * 2);        // aliased as h1 later
    __bf16* k     = (__bf16*)alloc((size_t)BS * H * 2);        // aliased as ctx later
    __bf16* vT    = (__bf16*)alloc((size_t)B * H * S * 2);
    float*  scor  = (float*) alloc((size_t)B * S * S * 4);     // aliased as fpre later
    __bf16* P     = (__bf16*)alloc((size_t)B * S * S * 2);
    float*  a     = (float*) alloc((size_t)BS * D * 4);
    float*  y1    = (float*) alloc((size_t)BS * D * 4);
    __bf16* y1bf  = (__bf16*)alloc((size_t)BS * D * 2);
    __bf16* h1    = q;      // q dead after scores GEMM
    __bf16* ctx   = k;      // k dead after scores GEMM
    float*  fpre  = scor;   // scores dead after softmax

    // --- stats must be zero each call (atomics accumulate) ---
    te_zero<<<1, 64, 0, stream>>>(stats, 64);

    // --- weight convert/transpose to bf16 [N,K] ---
    {
        const int nDH = D * H;
        const int g = (nDH + 255) / 256;
        te_cvt_t<<<g, 256, 0, stream>>>(Wq, WqT, D, H);   // [D,H] -> [H,D]
        te_cvt_t<<<g, 256, 0, stream>>>(Wk, WkT, D, H);
        te_cvt_t<<<g, 256, 0, stream>>>(Wv, WvT, D, H);
        te_cvt_t<<<g, 256, 0, stream>>>(Wo, WoT, H, D);   // [H,D] -> [D,H]
        te_cvt_t<<<g, 256, 0, stream>>>(W1, W1T, D, H);
        te_cvt_t<<<g, 256, 0, stream>>>(W2, W2T, H, D);
    }

    // --- embedding + positional encoding ---
    te_embed<<<(int)BS, 256, 0, stream>>>(x, emb, y, ybf);

    // --- Q = y@Wq + bq   (bf16 out, row-major) ---
    te_gemm<1, true><<<dim3(H / 64, (int)(BS / 256), 1), 256, 0, stream>>>(
        ybf, 0, D, WqT, 0, D, (void*)q, 0, H, bq, D);
    // --- K ---
    te_gemm<1, true><<<dim3(H / 64, (int)(BS / 256), 1), 256, 0, stream>>>(
        ybf, 0, D, WkT, 0, D, (void*)k, 0, H, bk, D);
    // --- V, stored transposed per batch: vT[b][h][s] ---
    te_gemm<2, true><<<dim3(H / 64, S / 256, B), 256, 0, stream>>>(
        ybf, (long)S * D, D, WvT, 0, D, (void*)vT, (long)H * S, S, bv, D);

    // --- scores[b,s,t] = q . k (unscaled), f32 out ---
    te_gemm<0, false><<<dim3(S / 64, S / 256, B), 256, 0, stream>>>(
        q, (long)S * H, H, k, (long)S * H, H, (void*)scor, (long)S * S, S,
        nullptr, H);

    // --- softmax rows -> bf16 probabilities ---
    te_softmax<<<(int)BS, 256, 0, stream>>>(scor, P);

    // --- ctx[b,s,h] = P @ v  (Bt = vT[b] is [H,S]) ---
    te_gemm<1, false><<<dim3(H / 64, S / 256, B), 256, 0, stream>>>(
        P, (long)S * S, S, vT, (long)H * S, S, (void*)ctx, (long)S * H, H,
        nullptr, S);

    // --- a = ctx @ Wo + bo  (f32) ---
    te_gemm<0, true><<<dim3(D / 64, (int)(BS / 256), 1), 256, 0, stream>>>(
        ctx, 0, H, WoT, 0, H, (void*)a, 0, D, bo, H);

    // --- ln1(y + a) -> y1 (f32) + y1bf (bf16) ---
    te_ln_stats<false><<<dim3(64, B), 256, 0, stream>>>(y, a, stats);
    te_ln_apply<false, true><<<dim3(64, B), 256, 0, stream>>>(
        y, a, ln1w, ln1b, stats, y1, y1bf);

    // --- h1 = y1 @ W1 + b1 (bf16) ---
    te_gemm<1, true><<<dim3(H / 64, (int)(BS / 256), 1), 256, 0, stream>>>(
        y1bf, 0, D, W1T, 0, D, (void*)h1, 0, H, b1, D);

    // --- fpre = h1 @ W2 + b2 (f32); GELU fused into ln2 input ---
    te_gemm<0, true><<<dim3(D / 64, (int)(BS / 256), 1), 256, 0, stream>>>(
        h1, 0, H, W2T, 0, H, (void*)fpre, 0, D, b2, H);

    // --- ln2(y1 + gelu(fpre)) -> d_out (f32) ---
    te_ln_stats<true><<<dim3(64, B), 256, 0, stream>>>(y1, fpre, stats + 16);
    te_ln_apply<true, false><<<dim3(64, B), 256, 0, stream>>>(
        y1, fpre, ln2w, ln2b, stats + 16, (float*)d_out, nullptr);
}
